// VRNN_68247030334234
// MI455X (gfx1250) — matrix operand order repaired and checked by reading, not compiled, not run on hardware
//
#include <hip/hip_runtime.h>
#include <hip/hip_bf16.h>
#include <math.h>

// ---------------------------------------------------------------------------
// VRNN on MI455X (gfx1250): LDS-free bf16 WMMA GEMM pipeline.
// Weights pre-transposed to bf16 [N][K]; activations bf16; fragments loaded
// with global_load_b128 (explicit addrspace(1)) from L2; per-segment loops
// with pure pointer increments; batched B loads stagger loadcnt waits.
// ---------------------------------------------------------------------------

typedef __attribute__((ext_vector_type(16))) __bf16 v16bf;
typedef __attribute__((ext_vector_type(8)))  __bf16 v8bf;
typedef __attribute__((ext_vector_type(8)))  float  v8f;

// global-address-space views (force global_load_b128 / global_store, not flat)
typedef const v8bf __attribute__((address_space(1)))* gv8bf_cptr;
typedef __bf16 __attribute__((address_space(1)))*     gbf16_ptr;

__device__ __forceinline__ v8bf gload8(const __bf16* p)
{
    return *(gv8bf_cptr)p;
}

#define BATCH 512
#define DIMD  256
#define DIML  256
#define KCL   16
#define TSEQ  64

#define ACT_NONE     0
#define ACT_RELU     1
#define ACT_TANH     2
#define ACT_RELU_EPS 3   // relu(x) + 1e-4 (alpha layers)

// ---------------------------------------------------------------------------
// out[512 x N] (bf16) = act( concat(A0,A1,A2)[512 x Ktot] (bf16) @ Wb^T + b )
// Wb is [N][Ktot] bf16 (pre-transposed).
// Block = 128 threads = 4 waves; wave tile = 16 rows x 64 cols.
// Grid = (BATCH/64, ceil(N/64)).
// ---------------------------------------------------------------------------
__global__ __launch_bounds__(128) void gemm_wmma_kernel(
    const __bf16* __restrict__ A0, int w0, int ld0,
    const __bf16* __restrict__ A1, int w1, int ld1,
    const __bf16* __restrict__ A2, int w2, int ld2,
    const __bf16* __restrict__ Wb,                 // [N][Ktot] bf16
    const float*  __restrict__ bias,
    __bf16* __restrict__ out, int N, int Ktot, int act)
{
    const int tid    = threadIdx.x;
    const int wv     = tid >> 5;
    const int lane   = tid & 31;
    const int half   = lane >> 4;      // K-half selector (ISA 7.12.2)
    const int m      = lane & 15;      // row (A) or col (B/C) within 16
    const int mBlock = blockIdx.x * 64;
    const int nBlock = blockIdx.y * 64;
    const int rowA   = mBlock + wv * 16 + m;

    // B-column base pointers (named, not an array, so addrspace inference and
    // register allocation stay clean). Clamped for narrow layers; stores are
    // guarded; EXEC stays all-ones for WMMA. Advance 32 elements per chunk.
    int n0 = nBlock + 0  + m; n0 = (n0 < N) ? n0 : (N - 1);
    int n1 = nBlock + 16 + m; n1 = (n1 < N) ? n1 : (N - 1);
    int n2 = nBlock + 32 + m; n2 = (n2 < N) ? n2 : (N - 1);
    int n3 = nBlock + 48 + m; n3 = (n3 < N) ? n3 : (N - 1);
    const __bf16* bp0 = Wb + (size_t)n0 * Ktot + half * 16;
    const __bf16* bp1 = Wb + (size_t)n1 * Ktot + half * 16;
    const __bf16* bp2 = Wb + (size_t)n2 * Ktot + half * 16;
    const __bf16* bp3 = Wb + (size_t)n3 * Ktot + half * 16;

    v8f acc0, acc1, acc2, acc3;
    #pragma unroll
    for (int j = 0; j < 8; ++j) { acc0[j] = 0.f; acc1[j] = 0.f; acc2[j] = 0.f; acc3[j] = 0.f; }

    // three sequential segment loops: base pointer computed once per segment,
    // then pure pointer increments (no per-chunk branches / 64-bit muls).
    const __bf16* segPtr[3] = {A0, A1, A2};
    const int     segW[3]   = {w0, w1, w2};
    const int     segLd[3]  = {ld0, ld1, ld2};

    for (int s = 0; s < 3; ++s) {
        const int wS = segW[s];
        if (wS == 0) continue;
        const __bf16* arow = segPtr[s] + (size_t)rowA * segLd[s] + half * 16;

        for (int kk = 0; kk < wS; kk += 32) {
            // ---- A fragment: two contiguous 16B global loads ----
            v8bf a0 = gload8(arow);
            v8bf a1 = gload8(arow + 8);
            v16bf afrag = __builtin_shufflevector(a0, a1,
                0, 1, 2, 3, 4, 5, 6, 7, 8, 9, 10, 11, 12, 13, 14, 15);

            // ---- all 4 B fragments loaded up-front (staggered waits) ----
            v8bf b0lo = gload8(bp0), b0hi = gload8(bp0 + 8);
            v8bf b1lo = gload8(bp1), b1hi = gload8(bp1 + 8);
            v8bf b2lo = gload8(bp2), b2hi = gload8(bp2 + 8);
            v8bf b3lo = gload8(bp3), b3hi = gload8(bp3 + 8);

            v16bf bf0 = __builtin_shufflevector(b0lo, b0hi,
                0, 1, 2, 3, 4, 5, 6, 7, 8, 9, 10, 11, 12, 13, 14, 15);
            v16bf bf1 = __builtin_shufflevector(b1lo, b1hi,
                0, 1, 2, 3, 4, 5, 6, 7, 8, 9, 10, 11, 12, 13, 14, 15);
            v16bf bf2 = __builtin_shufflevector(b2lo, b2hi,
                0, 1, 2, 3, 4, 5, 6, 7, 8, 9, 10, 11, 12, 13, 14, 15);
            v16bf bf3 = __builtin_shufflevector(b3lo, b3hi,
                0, 1, 2, 3, 4, 5, 6, 7, 8, 9, 10, 11, 12, 13, 14, 15);

            acc0 = __builtin_amdgcn_wmma_f32_16x16x32_bf16(
                false, afrag, false, bf0, (short)0, acc0, false, false);
            acc1 = __builtin_amdgcn_wmma_f32_16x16x32_bf16(
                false, afrag, false, bf1, (short)0, acc1, false, false);
            acc2 = __builtin_amdgcn_wmma_f32_16x16x32_bf16(
                false, afrag, false, bf2, (short)0, acc2, false, false);
            acc3 = __builtin_amdgcn_wmma_f32_16x16x32_bf16(
                false, afrag, false, bf3, (short)0, acc3, false, false);

            arow += 32;
            bp0 += 32; bp1 += 32; bp2 += 32; bp3 += 32;
        }
    }

    // ---- epilogue: bias + activation, bf16 global store ----
    v8f accs[4] = {acc0, acc1, acc2, acc3};
    #pragma unroll
    for (int st = 0; st < 4; ++st) {
        int ng = nBlock + st * 16 + m;
        if (ng >= N) continue;
        float bs = bias[ng];
        #pragma unroll
        for (int r = 0; r < 8; ++r) {
            int row = mBlock + wv * 16 + half * 8 + r;
            float v = accs[st][r] + bs;
            if (act == ACT_RELU)          v = fmaxf(v, 0.0f);
            else if (act == ACT_TANH)     v = tanhf(v);
            else if (act == ACT_RELU_EPS) v = fmaxf(v, 0.0f) + 1e-4f;
            ((gbf16_ptr)out)[(size_t)row * N + ng] = (__bf16)v;
        }
    }
}

// ---------------------------------------------------------------------------
// One-time conversion kernels (weights constant across timesteps)
// ---------------------------------------------------------------------------
__global__ void f32_to_bf16_kernel(const float* __restrict__ in,
                                   __bf16* __restrict__ out, int n)
{
    int i = blockIdx.x * blockDim.x + threadIdx.x;
    if (i < n) out[i] = (__bf16)in[i];
}

// Wb[n*K + k] = W[k*N + n]  (fp32 row-major [K][N] -> bf16 [N][K])
__global__ void transpose_w_kernel(const float* __restrict__ W,
                                   __bf16* __restrict__ Wb, int K, int N)
{
    int idx = blockIdx.x * blockDim.x + threadIdx.x;
    if (idx >= K * N) return;
    int n = idx / K;
    int k = idx - n * K;
    Wb[(size_t)n * K + k] = (__bf16)W[(size_t)k * N + n];
}

__global__ void zero_bf16_kernel(__bf16* __restrict__ p, int n)
{
    int i = blockIdx.x * blockDim.x + threadIdx.x;
    if (i < n) p[i] = (__bf16)0.0f;
}

__global__ void zero_f32_kernel(float* __restrict__ p, int n)
{
    int i = blockIdx.x * blockDim.x + threadIdx.x;
    if (i < n) p[i] = 0.0f;
}

// ---------------------------------------------------------------------------
// Dirichlet inverse-transform sample: pi = normalize(gamma)
// ---------------------------------------------------------------------------
__global__ void pi_kernel(const __bf16* __restrict__ alpha,
                          const float* __restrict__ u,
                          float* __restrict__ pi)
{
    int b = blockIdx.x * blockDim.x + threadIdx.x;
    if (b >= BATCH) return;
    float g[KCL];
    float s = 0.0f;
    #pragma unroll
    for (int k = 0; k < KCL; ++k) {
        float a  = (float)alpha[b * KCL + k];
        float uu = u[b * KCL + k];
        float gg = powf(a * uu * expf(lgammaf(a)), 1.0f / a);
        g[k] = gg;
        s += gg;
    }
    float inv = 1.0f / (s + 1e-8f);
    #pragma unroll
    for (int k = 0; k < KCL; ++k) pi[b * KCL + k] = g[k] * inv;
}

// z = pi @ clus_means  (512x16 @ 16x256), writes bf16
__global__ void z_kernel(const float* __restrict__ pi,
                         const float* __restrict__ clus,
                         __bf16* __restrict__ z)
{
    int idx = blockIdx.x * blockDim.x + threadIdx.x;
    if (idx >= BATCH * DIML) return;
    int b = idx >> 8;
    int l = idx & 255;
    float s = 0.0f;
    #pragma unroll
    for (int k = 0; k < KCL; ++k) s += pi[b * KCL + k] * clus[k * DIML + l];
    z[idx] = (__bf16)s;
}

// recon = 0.5*sum( logvar + (x-mu)^2 * exp(-logvar) )
__global__ void recon_kernel(const float* __restrict__ xt, int ldx,
                             const __bf16* __restrict__ dec,
                             float* __restrict__ loss)
{
    int idx = blockIdx.x * blockDim.x + threadIdx.x;
    float v = 0.0f;
    if (idx < BATCH * DIMD) {
        int b = idx >> 8;
        int d = idx & 255;
        float mu = (float)dec[b * (2 * DIMD) + d];
        float lv = (float)dec[b * (2 * DIMD) + DIMD + d];
        float x  = xt[(size_t)b * ldx + d];
        float df = x - mu;
        v = lv + df * df * expf(-lv);
    }
    for (int o = 16; o > 0; o >>= 1) v += __shfl_down(v, o, 32);
    if ((threadIdx.x & 31) == 0) atomicAdd(loss, 0.5f * v);
}

__device__ __forceinline__ float digammaf_dev(float x)
{
    float r = 0.0f;
    while (x < 6.0f) { r -= 1.0f / x; x += 1.0f; }
    float inv  = 1.0f / x;
    float inv2 = inv * inv;
    r += logf(x) - 0.5f * inv
       - inv2 * (1.0f / 12.0f - inv2 * (1.0f / 120.0f - inv2 * (1.0f / 252.0f)));
    return r;
}

// KL( Dir(aq) || Dir(ap) ) summed over batch rows
__global__ void kl_kernel(const __bf16* __restrict__ aq,
                          const __bf16* __restrict__ ap,
                          float* __restrict__ loss)
{
    int b = blockIdx.x * blockDim.x + threadIdx.x;
    float kl = 0.0f;
    if (b < BATCH) {
        float a0 = 0.0f, b0 = 0.0f;
        #pragma unroll
        for (int k = 0; k < KCL; ++k) {
            a0 += (float)aq[b * KCL + k];
            b0 += (float)ap[b * KCL + k];
        }
        float dga0 = digammaf_dev(a0);
        kl = lgammaf(a0) - lgammaf(b0);
        #pragma unroll
        for (int k = 0; k < KCL; ++k) {
            float a  = (float)aq[b * KCL + k];
            float bb = (float)ap[b * KCL + k];
            kl += -lgammaf(a) + lgammaf(bb) + (a - bb) * (digammaf_dev(a) - dga0);
        }
    }
    for (int o = 16; o > 0; o >>= 1) kl += __shfl_down(kl, o, 32);
    if ((threadIdx.x & 31) == 0) atomicAdd(loss, kl);
}

__global__ void finish_kernel(const float* __restrict__ loss, float* __restrict__ out)
{
    if (threadIdx.x == 0 && blockIdx.x == 0) out[0] = loss[0] / (float)BATCH;
}

// ---------------------------------------------------------------------------
// Host orchestration
// ---------------------------------------------------------------------------
extern "C" void kernel_launch(void* const* d_in, const int* in_sizes, int n_in,
                              void* d_out, int out_size, void* d_ws, size_t ws_size,
                              hipStream_t stream)
{
    (void)out_size; (void)ws_size;
    const float* P[64];
    int cnt = (n_in < 64) ? n_in : 64;
    for (int i = 0; i < cnt; ++i) P[i] = (const float*)d_in[i];

    // Detect flattening order: insertion (x first) vs jax-sorted.
    const bool insertion = (in_sizes[0] == BATCH * TSEQ * DIMD);

    const float *x, *u, *clus;
    // MLP order: 0=phi_x 1=phi_z 2=encoder 3=decoder 4=prior 5=rnn
    const float *MW[6][3], *MB[6][3];
    {
        const int* base;
        static const int baseIns[6]  = {2, 8, 14, 20, 26, 32};
        static const int baseJax[6]  = {13, 19, 7, 1, 25, 31};
        if (insertion) { x = P[0]; u = P[1]; clus = P[38]; base = baseIns; }
        else           { clus = P[0]; u = P[37]; x = P[38]; base = baseJax; }
        for (int mi = 0; mi < 6; ++mi)
            for (int j = 0; j < 3; ++j) {
                MW[mi][j] = P[base[mi] + 2 * j];
                MB[mi][j] = P[base[mi] + 2 * j + 1];
            }
    }

    // layer shapes (K_in, N_out) per MLP
    static const int KK[6][3] = {
        {256, 512, 512}, {256, 512, 512}, {512, 512, 512},
        {512, 512, 512}, {256, 512, 512}, {768, 512, 512}};
    static const int NN[6][3] = {
        {512, 512, 256}, {512, 512, 256}, {512, 512, 16},
        {512, 512, 512}, {512, 512, 16},  {512, 512, 256}};

    // ---- workspace carve-up (256B aligned) ----
    char* wsb = (char*)d_ws;
    size_t off = 0;
    auto carveB = [&](size_t bytes) {
        void* p = wsb + off;
        off += (bytes + 255) & ~(size_t)255;
        return p;
    };
    float*  loss = (float*)carveB(sizeof(float));
    __bf16* xbf  = (__bf16*)carveB((size_t)BATCH * TSEQ * DIMD * 2);
    __bf16* Wb[6][3];
    for (int mi = 0; mi < 6; ++mi)
        for (int j = 0; j < 3; ++j)
            Wb[mi][j] = (__bf16*)carveB((size_t)KK[mi][j] * NN[mi][j] * 2);
    __bf16* hA  = (__bf16*)carveB((size_t)BATCH * DIML * 2);
    __bf16* hB  = (__bf16*)carveB((size_t)BATCH * DIML * 2);
    __bf16* phx = (__bf16*)carveB((size_t)BATCH * DIML * 2);
    __bf16* phz = (__bf16*)carveB((size_t)BATCH * DIML * 2);
    __bf16* zb  = (__bf16*)carveB((size_t)BATCH * DIML * 2);
    __bf16* t1  = (__bf16*)carveB((size_t)BATCH * 512 * 2);
    __bf16* t2  = (__bf16*)carveB((size_t)BATCH * 512 * 2);
    __bf16* dec = (__bf16*)carveB((size_t)BATCH * 512 * 2);
    __bf16* aq  = (__bf16*)carveB((size_t)BATCH * KCL * 2);
    __bf16* ap  = (__bf16*)carveB((size_t)BATCH * KCL * 2);
    float*  pib = (float*)carveB((size_t)BATCH * KCL * 4);

    // ---- one-time conversions ----
    {
        int nx = BATCH * TSEQ * DIMD;
        f32_to_bf16_kernel<<<(nx + 255) / 256, 256, 0, stream>>>(x, xbf, nx);
        for (int mi = 0; mi < 6; ++mi)
            for (int j = 0; j < 3; ++j) {
                int ne = KK[mi][j] * NN[mi][j];
                transpose_w_kernel<<<(ne + 255) / 256, 256, 0, stream>>>(
                    MW[mi][j], Wb[mi][j], KK[mi][j], NN[mi][j]);
            }
    }
    zero_f32_kernel<<<1, 64, 0, stream>>>(loss, 1);
    zero_bf16_kernel<<<(BATCH * DIML + 255) / 256, 256, 0, stream>>>(hA, BATCH * DIML);

    auto gemm = [&](const __bf16* A0, int w0, int l0,
                    const __bf16* A1, int w1, int l1,
                    const __bf16* A2, int w2, int l2,
                    const __bf16* W, const float* B, __bf16* O, int N, int act) {
        dim3 grid(BATCH / 64, (N + 63) / 64);
        int Kt = w0 + w1 + w2;
        gemm_wmma_kernel<<<grid, 128, 0, stream>>>(A0, w0, l0, A1, w1, l1, A2, w2, l2,
                                                   W, B, O, N, Kt, act);
    };

    __bf16* h    = hA;
    __bf16* hNew = hB;

    for (int t = 0; t < TSEQ; ++t) {
        const __bf16* xt  = xbf + (size_t)t * DIMD;   // row stride T*D
        const float*  xtf = x + (size_t)t * DIMD;
        const int     ldx = TSEQ * DIMD;
        const float*  ut  = u + (size_t)t * BATCH * KCL;

        // phi_x(x_t) -> phx
        gemm(xt, DIMD, ldx, nullptr, 0, 0, nullptr, 0, 0, Wb[0][0], MB[0][0], t1, 512, ACT_RELU);
        gemm(t1, 512, 512, nullptr, 0, 0, nullptr, 0, 0, Wb[0][1], MB[0][1], t2, 512, ACT_RELU);
        gemm(t2, 512, 512, nullptr, 0, 0, nullptr, 0, 0, Wb[0][2], MB[0][2], phx, DIML, ACT_TANH);

        // encoder([phx, h]) -> alpha_q (relu + eps)
        gemm(phx, DIML, DIML, h, DIML, DIML, nullptr, 0, 0, Wb[2][0], MB[2][0], t1, 512, ACT_RELU);
        gemm(t1, 512, 512, nullptr, 0, 0, nullptr, 0, 0, Wb[2][1], MB[2][1], t2, 512, ACT_RELU);
        gemm(t2, 512, 512, nullptr, 0, 0, nullptr, 0, 0, Wb[2][2], MB[2][2], aq, KCL, ACT_RELU_EPS);

        // pi = dirichlet_sample(alpha_q, u_t); z = pi @ clus_means
        pi_kernel<<<(BATCH + 255) / 256, 256, 0, stream>>>(aq, ut, pib);
        z_kernel<<<(BATCH * DIML + 255) / 256, 256, 0, stream>>>(pib, clus, zb);

        // phi_z(z) -> phz
        gemm(zb, DIML, DIML, nullptr, 0, 0, nullptr, 0, 0, Wb[1][0], MB[1][0], t1, 512, ACT_RELU);
        gemm(t1, 512, 512, nullptr, 0, 0, nullptr, 0, 0, Wb[1][1], MB[1][1], t2, 512, ACT_RELU);
        gemm(t2, 512, 512, nullptr, 0, 0, nullptr, 0, 0, Wb[1][2], MB[1][2], phz, DIML, ACT_TANH);

        // decoder([phz, h]) -> dec; recon loss
        gemm(phz, DIML, DIML, h, DIML, DIML, nullptr, 0, 0, Wb[3][0], MB[3][0], t1, 512, ACT_RELU);
        gemm(t1, 512, 512, nullptr, 0, 0, nullptr, 0, 0, Wb[3][1], MB[3][1], t2, 512, ACT_RELU);
        gemm(t2, 512, 512, nullptr, 0, 0, nullptr, 0, 0, Wb[3][2], MB[3][2], dec, 2 * DIMD, ACT_TANH);
        recon_kernel<<<(BATCH * DIMD + 255) / 256, 256, 0, stream>>>(xtf, ldx, dec, loss);

        // prior(h) -> alpha_p; KL
        gemm(h, DIML, DIML, nullptr, 0, 0, nullptr, 0, 0, Wb[4][0], MB[4][0], t1, 512, ACT_RELU);
        gemm(t1, 512, 512, nullptr, 0, 0, nullptr, 0, 0, Wb[4][1], MB[4][1], t2, 512, ACT_RELU);
        gemm(t2, 512, 512, nullptr, 0, 0, nullptr, 0, 0, Wb[4][2], MB[4][2], ap, KCL, ACT_RELU_EPS);
        kl_kernel<<<(BATCH + 255) / 256, 256, 0, stream>>>(aq, ap, loss);

        // rnn([phx, phz, h]) -> h_new
        gemm(phx, DIML, DIML, phz, DIML, DIML, h, DIML, DIML, Wb[5][0], MB[5][0], t1, 512, ACT_RELU);
        gemm(t1, 512, 512, nullptr, 0, 0, nullptr, 0, 0, Wb[5][1], MB[5][1], t2, 512, ACT_RELU);
        gemm(t2, 512, 512, nullptr, 0, 0, nullptr, 0, 0, Wb[5][2], MB[5][2], hNew, DIML, ACT_TANH);

        __bf16* tmp = h; h = hNew; hNew = tmp;
    }

    finish_kernel<<<1, 64, 0, stream>>>(loss, (float*)d_out);
}
